// Attention_63333587746960
// MI455X (gfx1250) — matrix operand order repaired and checked by reading, not compiled, not run on hardware
//
#include <hip/hip_runtime.h>
#include <hip/hip_bf16.h>

// ---------------------------------------------------------------------------
// MI455X (gfx1250) multi-head attention:
//   f32->f16 convert  ->  3x WMMA GEMM (QKV)  ->  flash attention  ->  O GEMM
// Matmuls: v_wmma_f32_16x16x32_f16 (wave32). Tile staging: async global->LDS
// (ASYNCcnt) with double buffering when the toolchain exposes the builtin.
// ---------------------------------------------------------------------------

typedef _Float16 f16;
typedef __attribute__((ext_vector_type(16))) _Float16 v16h;
typedef __attribute__((ext_vector_type(8)))  _Float16 v8h;
typedef __attribute__((ext_vector_type(4)))  _Float16 v4h;
typedef __attribute__((ext_vector_type(8)))  float    v8f;
typedef __attribute__((ext_vector_type(4)))  int      v4i;

#define D_MODEL 1024
#define SEQ     2048
#define NB      4
#define NH      16
#define HD      64
#define MTOT    (NB * SEQ)          // 8192 rows

#define AS1 __attribute__((address_space(1)))
#define AS3 __attribute__((address_space(3)))

#if __has_builtin(__builtin_amdgcn_global_load_async_to_lds_b128)
#define USE_ASYNC_LDS 1
#endif

// 16-byte global -> LDS copy (per lane). Async path uses the CDNA5
// GLOBAL_LOAD_ASYNC_TO_LDS_B128 instruction (tracked by ASYNCcnt).
// Builtin params are v4i pointers: addrspace(1) src, addrspace(3) dst.
__device__ __forceinline__ void cp16_to_lds(f16* lds_dst, const f16* gsrc) {
#if defined(USE_ASYNC_LDS)
  __builtin_amdgcn_global_load_async_to_lds_b128(
      (AS1 v4i*)(size_t)gsrc,
      (AS3 v4i*)(unsigned)(size_t)lds_dst, 0, 0);
#else
  *(v8h*)lds_dst = *(const v8h*)gsrc;
#endif
}

__device__ __forceinline__ void wait_cp() {
#if defined(USE_ASYNC_LDS)
#if __has_builtin(__builtin_amdgcn_s_wait_asynccnt)
  __builtin_amdgcn_s_wait_asynccnt(0);
#else
  asm volatile("s_wait_asynccnt 0" ::: "memory");
#endif
#endif
}

__device__ __forceinline__ v8f wmma16(v16h a, v16h b, v8f c) {
  // (neg_a, A, neg_b, B, c_mod, C, reuse_a, reuse_b)
  return __builtin_amdgcn_wmma_f32_16x16x32_f16(false, a, false, b,
                                                (short)0, c, false, false);
}

// A-matrix fragment, 16x32 f16 (ISA 7.12.2 "16-bit A-Matrix 16x32"):
//   lanes 0-15 : row = lane,     K = {0..7, 16..23}
//   lanes 16-31: row = lane-16,  K = {8..15, 24..31}
__device__ __forceinline__ v16h load_frag_a(const f16* base, int stride) {
  const int lane = threadIdx.x & 31;
  const int r    = lane & 15;
  const int hf   = lane >> 4;
  const f16* p = base + r * stride + hf * 8;
  v8h lo = *(const v8h*)(p);
  v8h hi = *(const v8h*)(p + 16);
  v16h out;
#pragma unroll
  for (int i = 0; i < 8; ++i) { out[i] = lo[i]; out[i + 8] = hi[i]; }
  return out;
}

// B-matrix fragment, 32x16 f16 (ISA B layout):
//   lanes 0-15 : col = lane,     K = {0..7, 16..23}
//   lanes 16-31: col = lane-16,  K = {8..15, 24..31}
// Source element (k, col) at base[col*stride + k].
__device__ __forceinline__ v16h load_frag_b(const f16* base, int stride) {
  const int lane = threadIdx.x & 31;
  const int c    = lane & 15;
  const int kb   = (lane >> 4) * 8;
  const f16* p = base + c * stride + kb;
  v8h lo = *(const v8h*)(p);
  v8h hi = *(const v8h*)(p + 16);
  v16h out;
#pragma unroll
  for (int i = 0; i < 8; ++i) { out[i] = lo[i]; out[i + 8] = hi[i]; }
  return out;
}

// ---------------------------------------------------------------------------
// f32 -> f16 elementwise convert (vectorized x4); bandwidth-trivial prepass.
// ---------------------------------------------------------------------------
__global__ __launch_bounds__(256) void cvt_kernel(const float4* __restrict__ src,
                                                  v4h* __restrict__ dst, int n4) {
  int i = blockIdx.x * 256 + threadIdx.x;
  if (i < n4) {
    float4 v = src[i];
    v4h o;
    o[0] = (f16)v.x; o[1] = (f16)v.y; o[2] = (f16)v.z; o[3] = (f16)v.w;
    dst[i] = o;
  }
}

// ---------------------------------------------------------------------------
// GEMM: Y[M,N] = A[M,K] * W[N,K]^T + bias[N]   (M=8192, N=K=1024, all f16 in)
// OUT_MODE 0: f32 out, row-major [M,N]                          (final O proj)
// OUT_MODE 1: f16 out, [B,H,S,hd], scaled by 1/sqrt(hd)         (Q)
// OUT_MODE 2: f16 out, [B,H,S,hd]                               (K)
// OUT_MODE 3: f16 out, [B,H,hd,S]  (transposed for PV B-frags)  (V)
// 256 threads = 8 waves; tile 128x128x(BK=64); wave subtile 64x32;
// double-buffered async staging -> 16 WMMAs per wave per barrier pair.
// ---------------------------------------------------------------------------
template <int OUT_MODE>
__global__ __launch_bounds__(256) void gemm_kernel(const f16* __restrict__ A,
                                                   const f16* __restrict__ Wh,
                                                   const float* __restrict__ bias,
                                                   void* __restrict__ Out) {
  __shared__ f16 As[2][128 * 64];   // [m][k]
  __shared__ f16 Bs[2][128 * 64];   // [n][k]

  const int tid = threadIdx.x;
  const int wid = tid >> 5;
  const int wm  = wid >> 2;          // 0..1 -> 64-row slice
  const int wn  = wid & 3;           // 0..3 -> 32-col slice
  const int m0  = blockIdx.y * 128;
  const int n0  = blockIdx.x * 128;

  auto stage = [&](int k0, int buf) {
#pragma unroll
    for (int t = 0; t < 4; ++t) {
      int idx = tid + t * 256;                 // 1024 chunks of 8 f16
      int row = idx >> 3, c = (idx & 7) * 8;
      cp16_to_lds(&As[buf][row * 64 + c], A  + (size_t)(m0 + row) * D_MODEL + k0 + c);
      cp16_to_lds(&Bs[buf][row * 64 + c], Wh + (size_t)(n0 + row) * D_MODEL + k0 + c);
    }
  };

  v8f acc[4][2] = {};

  stage(0, 0);
  wait_cp();
  __syncthreads();

  const int NSTAGE = D_MODEL / 64;   // 16
  for (int s = 0; s < NSTAGE; ++s) {
    const int buf = s & 1;
    if (s + 1 < NSTAGE) stage((s + 1) * 64, buf ^ 1);   // prefetch next tile
#pragma unroll
    for (int kk = 0; kk < 64; kk += 32) {
      v16h bf[2];
#pragma unroll
      for (int ni = 0; ni < 2; ++ni)
        bf[ni] = load_frag_b(&Bs[buf][(wn * 32 + ni * 16) * 64 + kk], 64);
#pragma unroll
      for (int mi = 0; mi < 4; ++mi) {
        v16h af = load_frag_a(&As[buf][(wm * 64 + mi * 16) * 64 + kk], 64);
#pragma unroll
        for (int ni = 0; ni < 2; ++ni)
          acc[mi][ni] = wmma16(af, bf[ni], acc[mi][ni]);
      }
    }
    wait_cp();          // next tile landed
    __syncthreads();    // everyone done reading current tile
  }

  // ---- epilogue: bias + scatter per C/D layout ----
  const int lane = tid & 31;
  const int hf   = lane >> 4;
  const int cn   = lane & 15;
#pragma unroll
  for (int ni = 0; ni < 2; ++ni) {
    const int n = n0 + wn * 32 + ni * 16 + cn;
    const float bv = bias[n];
#pragma unroll
    for (int mi = 0; mi < 4; ++mi) {
#pragma unroll
      for (int r = 0; r < 8; ++r) {
        const int m = m0 + wm * 64 + mi * 16 + hf * 8 + r;
        float v = acc[mi][ni][r] + bv;
        if (OUT_MODE == 0) {
          ((float*)Out)[(size_t)m * D_MODEL + n] = v;
        } else {
          const int b = m >> 11, s = m & (SEQ - 1);
          const int h = n >> 6, d = n & (HD - 1);
          f16* o = (f16*)Out;
          if (OUT_MODE == 1)
            o[(((size_t)(b * NH + h)) * SEQ + s) * HD + d] = (f16)(v * 0.125f);
          else if (OUT_MODE == 2)
            o[(((size_t)(b * NH + h)) * SEQ + s) * HD + d] = (f16)v;
          else // V transposed: [b,h,hd,s]
            o[(((size_t)(b * NH + h)) * HD + d) * SEQ + s] = (f16)v;
        }
      }
    }
  }
}

// ---------------------------------------------------------------------------
// Flash attention: grid (S/128, B*H); 8 waves/WG; wave owns 16 queries.
// K/V tiles (32 keys) staged once per WG into LDS (async, double-buffered)
// and shared by all 8 waves (8x less global traffic than per-wave loads).
// Online softmax via width-16 xor shuffles; P re-layout through per-wave LDS
// (in-order DS within a wave -> no barrier needed); 8 WMMAs / wave / step.
// ---------------------------------------------------------------------------
__global__ __launch_bounds__(256) void attn_kernel(const f16* __restrict__ Q,
                                                   const f16* __restrict__ Kf,
                                                   const f16* __restrict__ Vt,
                                                   f16* __restrict__ Ctx) {
  __shared__ f16 Klds[2][32 * 64];   // [key][hd]
  __shared__ f16 Vlds[2][64 * 32];   // [hd][key]
  __shared__ f16 P[8][16 * 32];      // per-wave probability tile

  const int tid  = threadIdx.x;
  const int wid  = tid >> 5;
  const int lane = tid & 31;
  const int hf   = lane >> 4;
  const int cn   = lane & 15;

  const int bh = blockIdx.y;          // b*16 + h
  const int b  = bh >> 4;
  const int h  = bh & 15;
  const int s0 = blockIdx.x * 128 + wid * 16;

  const f16* qb    = Q  + ((size_t)bh * SEQ + s0) * HD;
  const f16* kbase = Kf + (size_t)bh * SEQ * HD;
  const f16* vbase = Vt + (size_t)bh * HD * SEQ;

  auto stage = [&](int j0, int buf) {
    { int key = tid >> 3, c = (tid & 7) * 8;                 // 32x64 K tile
      cp16_to_lds(&Klds[buf][key * 64 + c], kbase + (size_t)(j0 + key) * HD + c); }
    { int d = tid >> 2, c = (tid & 3) * 8;                   // 64x32 V^T tile
      cp16_to_lds(&Vlds[buf][d * 32 + c], vbase + (size_t)d * SEQ + j0 + c); }
  };

  // Q fragments are loop-invariant (pre-scaled by 1/sqrt(hd)).
  const v16h qa0 = load_frag_a(qb, HD);        // hd 0..31
  const v16h qa1 = load_frag_a(qb + 32, HD);   // hd 32..63

  v8f o[4] = {};
  float mrow[8], drow[8];
#pragma unroll
  for (int r = 0; r < 8; ++r) { mrow[r] = -3.0e38f; drow[r] = 0.0f; }

  stage(0, 0);
  wait_cp();
  __syncthreads();

  for (int it = 0; it < SEQ / 32; ++it) {
    const int buf = it & 1;
    if (it + 1 < SEQ / 32) stage((it + 1) * 32, buf ^ 1);    // prefetch

    const f16* kb = &Klds[buf][0];
    v8f sc0 = {}, sc1 = {};
    sc0 = wmma16(qa0, load_frag_b(kb, 64), sc0);             // keys 0..15
    sc0 = wmma16(qa1, load_frag_b(kb + 32, 64), sc0);
    sc1 = wmma16(qa0, load_frag_b(kb + 16 * 64, 64), sc1);   // keys 16..31
    sc1 = wmma16(qa1, load_frag_b(kb + 16 * 64 + 32, 64), sc1);

    float fac[8];
#pragma unroll
    for (int r = 0; r < 8; ++r) {
      float v0 = sc0[r], v1 = sc1[r];
      float mx = fmaxf(v0, v1);
#pragma unroll
      for (int off = 8; off >= 1; off >>= 1)
        mx = fmaxf(mx, __shfl_xor(mx, off, 16));             // row max
      const float mnew = fmaxf(mrow[r], mx);
      const float sca  = __expf(mrow[r] - mnew);
      const float p0   = __expf(v0 - mnew);
      const float p1   = __expf(v1 - mnew);
      float ps = p0 + p1;
#pragma unroll
      for (int off = 8; off >= 1; off >>= 1)
        ps += __shfl_xor(ps, off, 16);                       // row sum
      drow[r] = drow[r] * sca + ps;
      mrow[r] = mnew;
      fac[r]  = sca;
      const int prow = r + hf * 8;                           // C-layout row
      P[wid][prow * 32 + cn]      = (f16)p0;
      P[wid][prow * 32 + 16 + cn] = (f16)p1;
    }
#pragma unroll
    for (int c = 0; c < 4; ++c)
#pragma unroll
      for (int r = 0; r < 8; ++r) o[c][r] *= fac[r];

    // Per-wave LDS: DS ops are in-order within a wave, no barrier needed.
    const v16h pa = load_frag_a(&P[wid][0], 32);             // 16q x 32k
#pragma unroll
    for (int c = 0; c < 4; ++c)
      o[c] = wmma16(pa, load_frag_b(&Vlds[buf][(c * 16) * 32], 32), o[c]);

    wait_cp();          // next K/V tile landed
    __syncthreads();    // all waves done with current tile
  }

  // normalize and store context as [b, s, h*64+d] f16 (row-major M x 1024)
#pragma unroll
  for (int r = 0; r < 8; ++r) {
    const float inv = 1.0f / drow[r];
    const int s = s0 + hf * 8 + r;
#pragma unroll
    for (int c = 0; c < 4; ++c) {
      const int d = c * 16 + cn;
      Ctx[((size_t)b * SEQ + s) * D_MODEL + h * HD + d] = (f16)(o[c][r] * inv);
    }
  }
}

// ---------------------------------------------------------------------------
extern "C" void kernel_launch(void* const* d_in, const int* in_sizes, int n_in,
                              void* d_out, int out_size, void* d_ws, size_t ws_size,
                              hipStream_t stream) {
  (void)in_sizes; (void)n_in; (void)out_size; (void)ws_size;
  const float* x  = (const float*)d_in[0];
  const float* Wq = (const float*)d_in[1];
  const float* Wk = (const float*)d_in[2];
  const float* Wv = (const float*)d_in[3];
  const float* Wo = (const float*)d_in[4];
  const float* bq = (const float*)d_in[5];
  const float* bk = (const float*)d_in[6];
  const float* bv = (const float*)d_in[7];
  const float* bo = (const float*)d_in[8];

  char* ws = (char*)d_ws;
  const size_t XSEG = (size_t)MTOT * D_MODEL * sizeof(f16);      // 16 MB
  const size_t WSEG = (size_t)D_MODEL * D_MODEL * sizeof(f16);   //  2 MB
  f16* xh  = (f16*)(ws);
  f16* Wqh = (f16*)(ws + XSEG);
  f16* Wkh = (f16*)(ws + XSEG + 1 * WSEG);
  f16* Wvh = (f16*)(ws + XSEG + 2 * WSEG);
  f16* Woh = (f16*)(ws + XSEG + 3 * WSEG);
  f16* Qw  = (f16*)(ws + XSEG + 4 * WSEG);
  f16* Kw  = (f16*)(ws + 2 * XSEG + 4 * WSEG);
  f16* Vw  = (f16*)(ws + 3 * XSEG + 4 * WSEG);
  f16* Ctx = (f16*)(ws + 4 * XSEG + 4 * WSEG);

  // f32 -> f16 prepass (x + 4 weight matrices)
  const int xn4 = MTOT * D_MODEL / 4;          // 2097152
  const int wn4 = D_MODEL * D_MODEL / 4;       //  262144
  cvt_kernel<<<xn4 / 256, 256, 0, stream>>>((const float4*)x, (v4h*)xh, xn4);
  cvt_kernel<<<wn4 / 256, 256, 0, stream>>>((const float4*)Wq, (v4h*)Wqh, wn4);
  cvt_kernel<<<wn4 / 256, 256, 0, stream>>>((const float4*)Wk, (v4h*)Wkh, wn4);
  cvt_kernel<<<wn4 / 256, 256, 0, stream>>>((const float4*)Wv, (v4h*)Wvh, wn4);
  cvt_kernel<<<wn4 / 256, 256, 0, stream>>>((const float4*)Wo, (v4h*)Woh, wn4);

  dim3 gg(D_MODEL / 128, MTOT / 128);   // (8, 64)
  gemm_kernel<1><<<gg, 256, 0, stream>>>(xh, Wqh, bq, (void*)Qw);
  gemm_kernel<2><<<gg, 256, 0, stream>>>(xh, Wkh, bk, (void*)Kw);
  gemm_kernel<3><<<gg, 256, 0, stream>>>(xh, Wvh, bv, (void*)Vw);

  dim3 ga(SEQ / 128, NB * NH);          // (16, 64)
  attn_kernel<<<ga, 256, 0, stream>>>(Qw, Kw, Vw, Ctx);

  gemm_kernel<0><<<gg, 256, 0, stream>>>(Ctx, Woh, bo, d_out);
}
